// DecoderLayer_34634616275159
// MI455X (gfx1250) — matrix operand order repaired and checked by reading, not compiled, not run on hardware
//
#include <hip/hip_runtime.h>
#include <math.h>

typedef __attribute__((ext_vector_type(16))) __bf16 v16bf;
typedef __attribute__((ext_vector_type(8)))  float v8f;

#define DM 512
#define LQ 25
#define BATCH 128
#define HID 25
#define GATES 100
#define TSTEPS 512

// ---------- helpers ----------
__device__ __forceinline__ unsigned short f2bf(float f) {
    unsigned int u = __float_as_uint(f);
    unsigned int r = u + 0x7FFFu + ((u >> 16) & 1u);   // round-to-nearest-even
    return (unsigned short)(r >> 16);
}
__device__ __forceinline__ float sigmf(float x) { return 1.0f / (1.0f + __expf(-x)); }

union Frag16 { v16bf v; uint4 q[2]; unsigned short u[16]; };

// ---------- WMMA GEMM: C[M,512] = A[M,512] @ W (+bias), W is [512,512] (or W^T if BT) ----------
// Block tile 128x64, 256 threads (8 waves), each wave owns a 16x64 strip.
// LDS layouts are K-contiguous per lane so fragments load as 2x ds_load_b128:
//   As[m][k]  (row stride 40 ushorts = 80B, 16B-aligned chunks, 20-bank stride)
//   BsT[n][k] (B stored transposed)
template<bool BT>
__global__ __launch_bounds__(256)
void gemm_bf16_wmma(const float* __restrict__ A, const float* __restrict__ W,
                    const float* __restrict__ bias, float* __restrict__ C, int M) {
    __shared__ unsigned short As[128][40];
    __shared__ unsigned short BsT[64][40];
    const int tid  = threadIdx.x;
    const int lane = tid & 31;
    const int wave = tid >> 5;
    const int m0 = blockIdx.y * 128;
    const int n0 = blockIdx.x * 64;

    v8f acc[4] = {};

    for (int k0 = 0; k0 < DM; k0 += 32) {
        // stage A tile 128x32 (fp32 -> bf16), global reads coalesced along k
        for (int idx = tid; idx < 128 * 32; idx += 256) {
            int r = idx >> 5, c = idx & 31;
            As[r][c] = f2bf(A[(size_t)(m0 + r) * DM + (k0 + c)]);
        }
        // stage B tile as BsT[n][k]; keep the *global* fast index coalesced
        if (!BT) {
            for (int idx = tid; idx < 32 * 64; idx += 256) {
                int kk = idx >> 6, n = idx & 63;                 // n fast -> coalesced
                BsT[n][kk] = f2bf(W[(size_t)(k0 + kk) * DM + (n0 + n)]);
            }
        } else {
            for (int idx = tid; idx < 32 * 64; idx += 256) {
                int n = idx >> 5, kk = idx & 31;                 // k fast -> coalesced
                BsT[n][kk] = f2bf(W[(size_t)(n0 + n) * DM + (k0 + kk)]);
            }
        }
        __syncthreads();

        // A fragment (ISA 16-bit A 16x32 layout): two contiguous 16B chunks
        Frag16 a;
        const int mrow = 16 * wave + (lane & 15);
        const int kb = (lane >> 4) * 8;
        a.q[0] = *reinterpret_cast<const uint4*>(&As[mrow][kb]);
        a.q[1] = *reinterpret_cast<const uint4*>(&As[mrow][16 + kb]);

        const int kh = (lane >> 4) * 8;
        #pragma unroll
        for (int j = 0; j < 4; ++j) {
            const int nrow = 16 * j + (lane & 15);
            Frag16 b;
            b.q[0] = *reinterpret_cast<const uint4*>(&BsT[nrow][kh]);
            b.q[1] = *reinterpret_cast<const uint4*>(&BsT[nrow][16 + kh]);
            acc[j] = __builtin_amdgcn_wmma_f32_16x16x32_bf16(
                false, a.v, false, b.v, (short)0, acc[j], false, false);
        }
        __syncthreads();
    }

    const int ncol = lane & 15;
    const int rbase = m0 + 16 * wave + 8 * (lane >> 4);
    #pragma unroll
    for (int j = 0; j < 4; ++j) {
        const int col = n0 + 16 * j + ncol;
        const float bv = bias[col];
        #pragma unroll
        for (int v = 0; v < 8; ++v)
            C[(size_t)(rbase + v) * DM + col] = acc[j][v] + bv;
    }
}

// ---------- attention: per (b,h) scores -> softmax -> context ----------
__global__ __launch_bounds__(256)
void attn_kernel(const float* __restrict__ Q, const float* __restrict__ Kk,
                 const float* __restrict__ V, float* __restrict__ O, int S) {
    __shared__ float sq[LQ][64];
    __shared__ float sk[96][64];
    __shared__ float ss[LQ][96];
    const int b = blockIdx.x >> 3;
    const int h = blockIdx.x & 7;
    const int tid = threadIdx.x;

    for (int idx = tid; idx < LQ * 64; idx += 256) {
        int l = idx >> 6, e = idx & 63;
        sq[l][e] = Q[((size_t)(b * LQ + l)) * DM + h * 64 + e];
    }
    for (int idx = tid; idx < S * 64; idx += 256) {
        int s = idx >> 6, e = idx & 63;
        sk[s][e] = Kk[((size_t)(b * S + s)) * DM + h * 64 + e];
    }
    __syncthreads();

    for (int idx = tid; idx < LQ * S; idx += 256) {
        int l = idx / S, s = idx % S;
        float acc = 0.f;
        #pragma unroll 8
        for (int e = 0; e < 64; ++e) acc += sq[l][e] * sk[s][e];
        ss[l][s] = acc * 0.125f;   // 1/sqrt(64)
    }
    __syncthreads();

    if (tid < LQ) {
        float mx = -1e30f;
        for (int s = 0; s < S; ++s) mx = fmaxf(mx, ss[tid][s]);
        float sum = 0.f;
        for (int s = 0; s < S; ++s) { float e = __expf(ss[tid][s] - mx); ss[tid][s] = e; sum += e; }
        float inv = 1.f / sum;
        for (int s = 0; s < S; ++s) ss[tid][s] *= inv;
    }
    __syncthreads();

    for (int idx = tid; idx < LQ * 64; idx += 256) {
        int l = idx >> 6, e = idx & 63;
        float acc = 0.f;
        for (int s = 0; s < S; ++s)
            acc += ss[l][s] * V[((size_t)(b * S + s)) * DM + h * 64 + e];
        O[((size_t)(b * LQ + l)) * DM + h * 64 + e] = acc;
    }
}

// ---------- fused residual add + LayerNorm (biased var, eps=1e-5) ----------
__global__ __launch_bounds__(256)
void add_ln_kernel(const float* __restrict__ X, const float* __restrict__ P,
                   const float* __restrict__ G, const float* __restrict__ Bb,
                   float* __restrict__ O) {
    __shared__ float buf[512];
    __shared__ float r1[256], r2[256];
    __shared__ float s_mean, s_inv;
    const int row = blockIdx.x, tid = threadIdx.x;
    const size_t base = (size_t)row * DM;
    float v0 = X[base + tid]       + P[base + tid];
    float v1 = X[base + tid + 256] + P[base + tid + 256];
    buf[tid] = v0; buf[tid + 256] = v1;
    r1[tid] = v0 + v1; r2[tid] = v0 * v0 + v1 * v1;
    __syncthreads();
    for (int s = 128; s > 0; s >>= 1) {
        if (tid < s) { r1[tid] += r1[tid + s]; r2[tid] += r2[tid + s]; }
        __syncthreads();
    }
    if (tid == 0) {
        float m = r1[0] * (1.f / 512.f);
        float var = r2[0] * (1.f / 512.f) - m * m;
        s_mean = m; s_inv = rsqrtf(var + 1e-5f);
    }
    __syncthreads();
    O[base + tid]       = (buf[tid]       - s_mean) * s_inv * G[tid]       + Bb[tid];
    O[base + tid + 256] = (buf[tid + 256] - s_mean) * s_inv * G[tid + 256] + Bb[tid + 256];
}

// ---------- transposes around LSTM ----------
__global__ void transpose_x_to_y(const float* __restrict__ X, float* __restrict__ Y) {
    // X:[B,25,512] -> Y:[B,512,25]
    int idx = blockIdx.x * 256 + threadIdx.x;
    if (idx >= BATCH * 512 * 25) return;
    int f = idx % 25, t = (idx / 25) % 512, b = idx / (25 * 512);
    Y[idx] = X[((size_t)b * 25 + f) * 512 + t];
}
__global__ void transpose_y_to_z(const float* __restrict__ Y, float* __restrict__ Z) {
    // Y:[B,512,25] -> Z:[B,25,512]
    int idx = blockIdx.x * 256 + threadIdx.x;
    if (idx >= BATCH * 512 * 25) return;
    int t = idx % 512, f = (idx / 512) % 25, b = idx / (512 * 25);
    Z[idx] = Y[((size_t)b * 512 + t) * 25 + f];
}

// ---------- one LSTM layer: persistent WG per batch element, weights+state in LDS ----------
__global__ __launch_bounds__(128)
void lstm_layer_kernel(const float* __restrict__ In, float* __restrict__ Out,
                       const float* __restrict__ Wih, const float* __restrict__ Whh,
                       const float* __restrict__ bih, const float* __restrict__ bhh) {
    __shared__ float sWih[GATES][HID + 1];
    __shared__ float sWhh[GATES][HID + 1];
    __shared__ float sb[GATES];
    __shared__ float sh[HID], sc[HID], sx[HID], sg[GATES];
    const int b = blockIdx.x, tid = threadIdx.x;
    for (int i = tid; i < GATES * HID; i += 128) {
        int g = i / HID, f = i % HID;
        sWih[g][f] = Wih[i]; sWhh[g][f] = Whh[i];
    }
    if (tid < GATES) sb[tid] = bih[tid] + bhh[tid];
    if (tid < HID) { sh[tid] = 0.f; sc[tid] = 0.f; }
    __syncthreads();
    const float* inb = In + (size_t)b * TSTEPS * HID;
    float* outb = Out + (size_t)b * TSTEPS * HID;
    for (int t = 0; t < TSTEPS; ++t) {
        if (tid < HID) sx[tid] = inb[t * HID + tid];
        __syncthreads();
        if (tid < GATES) {
            float acc = sb[tid];
            #pragma unroll
            for (int f = 0; f < HID; ++f)
                acc += sWih[tid][f] * sx[f] + sWhh[tid][f] * sh[f];
            sg[tid] = acc;
        }
        __syncthreads();
        if (tid < HID) {
            float ig = sg[tid], fg = sg[HID + tid], gg = sg[2 * HID + tid], og = sg[3 * HID + tid];
            float c = sigmf(fg) * sc[tid] + sigmf(ig) * tanhf(gg);
            float h = sigmf(og) * tanhf(c);
            sc[tid] = c; sh[tid] = h;
            outb[t * HID + tid] = h;
        }
        __syncthreads();
    }
}

// ---------- host orchestration ----------
extern "C" void kernel_launch(void* const* d_in, const int* in_sizes, int n_in,
                              void* d_out, int out_size, void* d_ws, size_t ws_size,
                              hipStream_t stream) {
    const float* x     = (const float*)d_in[0];
    const float* cross = (const float*)d_in[1];
    const float* Wq_s = (const float*)d_in[2];  const float* bq_s = (const float*)d_in[3];
    const float* Wk_s = (const float*)d_in[4];  const float* bk_s = (const float*)d_in[5];
    const float* Wv_s = (const float*)d_in[6];  const float* bv_s = (const float*)d_in[7];
    const float* Wo_s = (const float*)d_in[8];  const float* bo_s = (const float*)d_in[9];
    const float* Wq_c = (const float*)d_in[10]; const float* bq_c = (const float*)d_in[11];
    const float* Wk_c = (const float*)d_in[12]; const float* bk_c = (const float*)d_in[13];
    const float* Wv_c = (const float*)d_in[14]; const float* bv_c = (const float*)d_in[15];
    const float* Wo_c = (const float*)d_in[16]; const float* bo_c = (const float*)d_in[17];
    const float* g1 = (const float*)d_in[18];   const float* b1 = (const float*)d_in[19];
    const float* g2 = (const float*)d_in[20];   const float* b2 = (const float*)d_in[21];
    const float* g3 = (const float*)d_in[22];   const float* b3 = (const float*)d_in[23];
    const float* W_ih = (const float*)d_in[24];
    const float* W_hh = (const float*)d_in[25];
    const float* b_ih = (const float*)d_in[26];
    const float* b_hh = (const float*)d_in[27];
    const float* Wc = (const float*)d_in[28];   const float* bc = (const float*)d_in[29];
    float* out = (float*)d_out;

    const size_t R  = (size_t)BATCH * LQ * DM;   // 3200*512
    const size_t RC = (size_t)BATCH * 96 * DM;   // 12288*512
    float* ws = (float*)d_ws;
    float* bufQ  = ws;
    float* bufK  = bufQ + R;     // RC
    float* bufV  = bufK + RC;    // RC
    float* bufO  = bufV + RC;    // R
    float* bufP  = bufO + R;     // R
    float* bufX2 = bufP + R;     // R
    float* bufX3 = bufX2 + R;    // R
    float* Y0 = bufQ;            // alias: free after cross attention
    float* Y1 = bufO;            // alias: free after o-proj input consumed
    float* Zb = bufK;            // alias: free after cross attention

    const dim3 gblk(256);
    const dim3 gemm_grid_s(8, 3200 / 128);
    const dim3 gemm_grid_c(8, 12288 / 128);

    // self-attention
    gemm_bf16_wmma<false><<<gemm_grid_s, gblk, 0, stream>>>(x, Wq_s, bq_s, bufQ, 3200);
    gemm_bf16_wmma<false><<<gemm_grid_s, gblk, 0, stream>>>(x, Wk_s, bk_s, bufK, 3200);
    gemm_bf16_wmma<false><<<gemm_grid_s, gblk, 0, stream>>>(x, Wv_s, bv_s, bufV, 3200);
    attn_kernel<<<BATCH * 8, 256, 0, stream>>>(bufQ, bufK, bufV, bufO, 25);
    gemm_bf16_wmma<false><<<gemm_grid_s, gblk, 0, stream>>>(bufO, Wo_s, bo_s, bufP, 3200);
    add_ln_kernel<<<3200, 256, 0, stream>>>(x, bufP, g1, b1, bufX2);

    // cross-attention
    gemm_bf16_wmma<false><<<gemm_grid_s, gblk, 0, stream>>>(bufX2, Wq_c, bq_c, bufQ, 3200);
    gemm_bf16_wmma<false><<<gemm_grid_c, gblk, 0, stream>>>(cross, Wk_c, bk_c, bufK, 12288);
    gemm_bf16_wmma<false><<<gemm_grid_c, gblk, 0, stream>>>(cross, Wv_c, bv_c, bufV, 12288);
    attn_kernel<<<BATCH * 8, 256, 0, stream>>>(bufQ, bufK, bufV, bufO, 96);
    gemm_bf16_wmma<false><<<gemm_grid_s, gblk, 0, stream>>>(bufO, Wo_c, bo_c, bufP, 3200);
    add_ln_kernel<<<3200, 256, 0, stream>>>(bufX2, bufP, g2, b2, bufX3);

    // LSTM stack over feature axis
    const int NEL = BATCH * 512 * 25;
    transpose_x_to_y<<<(NEL + 255) / 256, 256, 0, stream>>>(bufX3, Y0);
    for (int l = 0; l < 32; ++l) {
        const float* in   = (l & 1) ? Y1 : Y0;
        float*       outy = (l & 1) ? Y0 : Y1;
        lstm_layer_kernel<<<BATCH, 128, 0, stream>>>(
            in, outy, W_ih + (size_t)l * GATES * HID, W_hh + (size_t)l * GATES * HID,
            b_ih + (size_t)l * GATES, b_hh + (size_t)l * GATES);
    }
    // 32 layers (even): final output in Y0
    transpose_y_to_z<<<(NEL + 255) / 256, 256, 0, stream>>>(Y0, Zb);

    // pointwise conv = Z @ Wc^T + bc, then residual + LN
    gemm_bf16_wmma<true><<<gemm_grid_s, gblk, 0, stream>>>(Zb, Wc, bc, bufP, 3200);
    add_ln_kernel<<<3200, 256, 0, stream>>>(bufX3, bufP, g3, b3, out);
}
